// RTGNCriticNet_72138270703873
// MI455X (gfx1250) — matrix operand order repaired and checked by reading, not compiled, not run on hardware
//
#include <hip/hip_runtime.h>
#include <cstdint>

#define N_NODES 15000
#define N_EDGES 30000
#define GDIM 64

typedef __attribute__((ext_vector_type(16))) __bf16 v16bf;
typedef __attribute__((ext_vector_type(8)))  float  v8f;

static __device__ __forceinline__ __bf16 f2bf(float f) {
  uint32_t u; __builtin_memcpy(&u, &f, 4);
  uint32_t r = u + 0x7FFFu + ((u >> 16) & 1u);
  uint16_t h = (uint16_t)(r >> 16);
  __bf16 o; __builtin_memcpy(&o, &h, 2); return o;
}
static __device__ __forceinline__ float bflo(uint32_t p) {
  uint32_t x = p << 16; float f; __builtin_memcpy(&f, &x, 4); return f;
}
static __device__ __forceinline__ float bfhi(uint32_t p) {
  uint32_t x = p & 0xFFFF0000u; float f; __builtin_memcpy(&f, &x, 4); return f;
}
static __device__ __forceinline__ float sigf(float x) { return 1.0f / (1.0f + __expf(-x)); }

// ---------------- generic zero ----------------
__global__ void zero_kernel(float* p, int n) {
  int i = blockIdx.x * blockDim.x + threadIdx.x;
  int stride = gridDim.x * blockDim.x;
  for (; i < n; i += stride) p[i] = 0.0f;
}

// ---------------- f32 -> bf16 weight conversion ----------------
__global__ void cvt_bf16_kernel(const float* __restrict__ s, __bf16* __restrict__ d, int n) {
  int i = blockIdx.x * blockDim.x + threadIdx.x;
  if (i < n) d[i] = f2bf(s[i]);
}
// 64x64 transpose convert: d[o][k] = s[k][o]  (root_w used as out @ root_w)
__global__ void cvt_bf16_t64_kernel(const float* __restrict__ s, __bf16* __restrict__ d) {
  int i = blockIdx.x * blockDim.x + threadIdx.x;
  if (i >= 64 * 64) return;
  int o = i >> 6, k = i & 63;
  d[i] = f2bf(s[k * 64 + o]);
}

// ---------------- lin0: out = relu(x @ lin0_w.T + b), x is [N,3] ----------------
__global__ void lin0_kernel(const float* __restrict__ x, const float* __restrict__ w,
                            const float* __restrict__ b, float* __restrict__ h,
                            __bf16* __restrict__ h_bf) {
  int idx = blockIdx.x * blockDim.x + threadIdx.x;
  if (idx >= N_NODES * GDIM) return;
  int n = idx >> 6, d = idx & 63;
  float s = b[d] + x[n * 3 + 0] * w[d * 3 + 0] + x[n * 3 + 1] * w[d * 3 + 1] +
            x[n * 3 + 2] * w[d * 3 + 2];
  s = fmaxf(s, 0.0f);
  h[idx] = s;
  h_bf[idx] = f2bf(s);
}

// ---------------- he = relu(edge_attr * e1_w + e1_b), ED=1 ----------------
__global__ void he_kernel(const float* __restrict__ ea, const float* __restrict__ w,
                          const float* __restrict__ b, __bf16* __restrict__ he_bf) {
  int idx = blockIdx.x * blockDim.x + threadIdx.x;
  if (idx >= N_EDGES * GDIM) return;
  int e = idx >> 6, d = idx & 63;
  float v = fmaxf(ea[e] * w[d] + b[d], 0.0f);
  he_bf[idx] = f2bf(v);
}

// ---------------- degree ----------------
__global__ void deg_kernel(const int* __restrict__ dst, float* __restrict__ deg) {
  int e = blockIdx.x * blockDim.x + threadIdx.x;
  if (e >= N_EDGES) return;
  atomicAdd(&deg[dst[e]], 1.0f);
}

// ---------------- WMMA GEMM: C[M,P] = A_bf16[M,64] @ Wb.T + bias ----------------
// Wb is pre-converted bf16 [P][64] row-major. One wave per 16x64 output strip
// (4 n-tiles, A fragment reused across 4 WMMAs per K-half -> 8 v_wmma per wave).
__global__ void gemm_wmma_k64(const __bf16* __restrict__ A, const __bf16* __restrict__ Wb,
                              const float* __restrict__ bias, float* __restrict__ Cf,
                              __bf16* __restrict__ Cb, int M, int P, int doRelu) {
  constexpr int NT = 4;
  int lane = threadIdx.x;
  int rlo = lane & 15, hi = lane >> 4;
  int m0 = blockIdx.x * 16;
  int n0 = blockIdx.y * (16 * NT);
  v8f acc[NT];
#pragma unroll
  for (int t = 0; t < NT; ++t)
#pragma unroll
    for (int r = 0; r < 8; ++r) acc[t][r] = 0.0f;

  int arow = m0 + rlo; if (arow >= M) arow = M - 1;
#pragma unroll
  for (int k0 = 0; k0 < 64; k0 += 32) {
    // A 16x32 layout (matches B): lane holds K = hi*16 + {0..15}
    const __bf16* ap = A + (size_t)arow * 64 + k0 + hi * 16;
    v16bf a;
    __builtin_memcpy(&a, ap, 32);                    // elements 0..15
    int kb = k0 + hi * 16;  // B 32x16 layout: K(j) = hi*16 + j, N = lane&15
#pragma unroll
    for (int t = 0; t < NT; ++t) {
      int col = n0 + t * 16 + rlo;
      v16bf b;
      __builtin_memcpy(&b, Wb + (size_t)col * 64 + kb, 32);  // 16 contiguous bf16
      acc[t] = __builtin_amdgcn_wmma_f32_16x16x32_bf16(false, a, false, b, (short)0,
                                                       acc[t], false, false);
    }
  }
#pragma unroll
  for (int t = 0; t < NT; ++t) {
    int col = n0 + t * 16 + rlo;
    float bi = bias ? bias[col] : 0.0f;
#pragma unroll
    for (int r = 0; r < 8; ++r) {
      int mr = m0 + r + hi * 8;  // C/D: row = r + hi*8, col = lane&15
      if (mr < M) {
        float v = acc[t][r] + bi;
        if (doRelu) v = fmaxf(v, 0.0f);
        size_t idx = (size_t)mr * P + col;
        if (Cf) Cf[idx] = v;
        if (Cb) Cb[idx] = f2bf(v);
      }
    }
  }
}

// ---------------- per-edge matvec + scatter: agg[dst] += out[src] @ We[e] ----------------
__global__ void msg_kernel(const int* __restrict__ src, const int* __restrict__ dst,
                           const __bf16* __restrict__ We, const float* __restrict__ h,
                           float* __restrict__ agg) {
  int wave = (blockIdx.x * blockDim.x + threadIdx.x) >> 5;
  int lane = threadIdx.x & 31;
  if (wave >= N_EDGES) return;
  int e = wave;
  int s = src[e], d = dst[e];
  const uint32_t* w = (const uint32_t*)(We + (size_t)e * 4096);  // [64 i][32 pairs]
  const float* xs = h + (size_t)s * 64;
  float a0 = 0.0f, a1 = 0.0f;
#pragma unroll 8
  for (int i = 0; i < 64; ++i) {
    float xi = xs[i];
    uint32_t p = w[i * 32 + lane];  // columns 2*lane, 2*lane+1 (coalesced 128B/row)
    a0 += xi * bflo(p);
    a1 += xi * bfhi(p);
  }
  atomicAdd(&agg[(size_t)d * 64 + 2 * lane], a0);
  atomicAdd(&agg[(size_t)d * 64 + 2 * lane + 1], a1);
}

// ---------------- m = relu(root_term + agg/denom); also reset agg ----------------
__global__ void combine_kernel(const float* __restrict__ deg, float* __restrict__ m,
                               __bf16* __restrict__ m_bf, float* __restrict__ agg) {
  int idx = blockIdx.x * blockDim.x + threadIdx.x;
  if (idx >= N_NODES * GDIM) return;
  int n = idx >> 6;
  float dg = deg[n];
  float den = dg > 1.0f ? dg : 1.0f;
  float v = m[idx] + agg[idx] / den;
  v = fmaxf(v, 0.0f);
  m[idx] = v;
  m_bf[idx] = f2bf(v);
  agg[idx] = 0.0f;
}

// ---------------- GRU elementwise ----------------
__global__ void gru_kernel(const float* __restrict__ gi, const float* __restrict__ gh,
                           float* __restrict__ h, __bf16* __restrict__ h_bf) {
  int idx = blockIdx.x * blockDim.x + threadIdx.x;
  if (idx >= N_NODES * GDIM) return;
  int n = idx >> 6, d = idx & 63;
  size_t base = (size_t)n * 192;
  float r = sigf(gi[base + d] + gh[base + d]);
  float z = sigf(gi[base + 64 + d] + gh[base + 64 + d]);
  float nn = tanhf(gi[base + 128 + d] + r * gh[base + 128 + d]);
  float hv = h[idx];
  float hn = (1.0f - z) * nn + z * hv;
  h[idx] = hn;
  h_bf[idx] = f2bf(hn);
}

// ---------------- Set2Set LSTM step (1 block, 256 thr); also zeros attn accumulators ----------------
__global__ void s2s_lstm_kernel(const float* __restrict__ wih, const float* __restrict__ whh,
                                const float* __restrict__ bih, const float* __restrict__ bhh,
                                const float* __restrict__ qstar, float* __restrict__ hs,
                                float* __restrict__ cs, float* __restrict__ r_acc,
                                float* __restrict__ sumexp) {
  __shared__ float q[128];
  __shared__ float hh[64];
  __shared__ float gate[256];
  int t = threadIdx.x;
  if (t < 128) q[t] = qstar[t];
  if (t < 64) hh[t] = hs[t];
  __syncthreads();
  float g = bih[t] + bhh[t];
  const float* wi = wih + (size_t)t * 128;
#pragma unroll 8
  for (int k = 0; k < 128; ++k) g += q[k] * wi[k];
  const float* wh = whh + (size_t)t * 64;
#pragma unroll 8
  for (int k = 0; k < 64; ++k) g += hh[k] * wh[k];
  gate[t] = g;
  __syncthreads();
  if (t < 64) {
    float i = gate[t], f = gate[64 + t], gg = gate[128 + t], o = gate[192 + t];
    float c2 = sigf(f) * cs[t] + sigf(i) * tanhf(gg);
    hs[t] = sigf(o) * tanhf(c2);
    cs[t] = c2;
    r_acc[t] = 0.0f;
  }
  if (t == 64) *sumexp = 0.0f;
}

// ---------------- attention: e = out@hs, softmax-weighted sum (no max-sub; |e|<=~64) ----------------
__global__ void attn_kernel(const float* __restrict__ h, const float* __restrict__ hs,
                            float* __restrict__ r_acc, float* __restrict__ sumexp) {
  __shared__ float lred[8 * 64];
  __shared__ float lsum[8];
  int lane = threadIdx.x & 31;
  int wid = threadIdx.x >> 5;
  float hs0 = hs[lane], hs1 = hs[lane + 32];
  float r0 = 0.0f, r1 = 0.0f, se = 0.0f;
  int gw = blockIdx.x * 8 + wid;
  int nw = gridDim.x * 8;
  for (int node = gw; node < N_NODES; node += nw) {
    const float* row = h + (size_t)node * 64;
    float x0 = row[lane], x1 = row[lane + 32];
    float p = x0 * hs0 + x1 * hs1;
#pragma unroll
    for (int s = 16; s; s >>= 1) p += __shfl_xor(p, s, 32);
    float ex = __expf(p);
    r0 += ex * x0;
    r1 += ex * x1;
    if (lane == 0) se += ex;
  }
  lred[wid * 64 + lane] = r0;
  lred[wid * 64 + 32 + lane] = r1;
  if (lane == 0) lsum[wid] = se;
  __syncthreads();
  if (threadIdx.x < 64) {
    float tsum = 0.0f;
#pragma unroll
    for (int w = 0; w < 8; ++w) tsum += lred[w * 64 + threadIdx.x];
    atomicAdd(&r_acc[threadIdx.x], tsum);
  }
  if (threadIdx.x == 64) {
    float tsum = 0.0f;
#pragma unroll
    for (int w = 0; w < 8; ++w) tsum += lsum[w];
    atomicAdd(sumexp, tsum);
  }
}

// ---------------- q_star = [hs, r_acc/sumexp] ----------------
__global__ void s2s_finalize_kernel(const float* __restrict__ hs, const float* __restrict__ r_acc,
                                    const float* __restrict__ sumexp, float* __restrict__ qstar) {
  int t = threadIdx.x;
  if (t >= 64) return;
  float inv = 1.0f / (*sumexp);
  qstar[t] = hs[t];
  qstar[64 + t] = r_acc[t] * inv;
}

// ---------------- head: mem LSTM (h=c=0) -> lin1 relu -> lin3 ----------------
__global__ void head_kernel(const float* __restrict__ qstar, const float* __restrict__ wih,
                            const float* __restrict__ bih, const float* __restrict__ bhh,
                            const float* __restrict__ l1w, const float* __restrict__ l1b,
                            const float* __restrict__ l3w, const float* __restrict__ l3b,
                            float* __restrict__ out) {
  __shared__ float q[128];
  __shared__ float gate[256];
  __shared__ float hx[64];
  __shared__ float ovec[64];
  int t = threadIdx.x;
  if (t < 128) q[t] = qstar[t];
  __syncthreads();
  float g = bih[t] + bhh[t];
  const float* wi = wih + (size_t)t * 128;
#pragma unroll 8
  for (int k = 0; k < 128; ++k) g += q[k] * wi[k];
  gate[t] = g;
  __syncthreads();
  if (t < 64) {
    float i = gate[t], f = gate[64 + t], gg = gate[128 + t], o = gate[192 + t];
    (void)f;  // c_prev == 0
    float c2 = sigf(i) * tanhf(gg);
    float hv = sigf(o) * tanhf(c2);
    hx[t] = hv;
    out[1 + t] = hv;       // hx
    out[65 + t] = c2;      // cx
  }
  __syncthreads();
  if (t < 64) {
    float s = l1b[t];
    const float* wr = l1w + (size_t)t * 64;
#pragma unroll 8
    for (int k = 0; k < 64; ++k) s += hx[k] * wr[k];
    ovec[t] = fmaxf(s, 0.0f);
  }
  __syncthreads();
  if (t == 0) {
    float v = l3b[0];
#pragma unroll 8
    for (int d = 0; d < 64; ++d) v += ovec[d] * l3w[d];
    out[0] = v;
  }
}

// ---------------- workspace layout (bytes) ----------------
#define WS_WE    ((size_t)0)                          // E*4096 bf16   = 245,760,000
#define WS_HE    ((size_t)245760000)                  // E*64   bf16   =   3,840,000
#define WS_H     ((size_t)249600000)                  // N*64   f32    =   3,840,000
#define WS_HBF   ((size_t)253440000)                  // N*64   bf16   =   1,920,000
#define WS_M     ((size_t)255360256)                  // N*64   f32
#define WS_MBF   ((size_t)259200256)                  // N*64   bf16
#define WS_AGG   ((size_t)261120512)                  // N*64   f32
#define WS_GI    ((size_t)264960512)                  // N*192  f32    =  11,520,000
#define WS_GH    ((size_t)276480512)                  // N*192  f32
#define WS_DEG   ((size_t)288000512)                  // N      f32
#define WS_SMALL ((size_t)288060672)                  // 512 f32: q[128] hs[64] cs[64] racc[64] sumexp[1]
#define WS_E2WBF ((size_t)288062720)                  // 4096*64 bf16 = 524,288
#define WS_GWIH  ((size_t)288587008)                  // 192*64 bf16  =  24,576
#define WS_GWHH  ((size_t)288611584)                  // 192*64 bf16  =  24,576
#define WS_ROOTW ((size_t)288636160)                  // 64*64  bf16 (transposed) = 8,192

extern "C" void kernel_launch(void* const* d_in, const int* in_sizes, int n_in,
                              void* d_out, int out_size, void* d_ws, size_t ws_size,
                              hipStream_t stream) {
  (void)in_sizes; (void)n_in; (void)out_size; (void)ws_size;
  const float* x      = (const float*)d_in[0];
  const int*   ei     = (const int*)  d_in[1];
  const float* eattr  = (const float*)d_in[2];
  const float* lin0_w = (const float*)d_in[3];
  const float* lin0_b = (const float*)d_in[4];
  const float* e1_w   = (const float*)d_in[5];
  const float* e1_b   = (const float*)d_in[6];
  const float* e2_w   = (const float*)d_in[7];
  const float* e2_b   = (const float*)d_in[8];
  const float* root_w = (const float*)d_in[9];
  const float* conv_b = (const float*)d_in[10];
  const float* gw_ih  = (const float*)d_in[11];
  const float* gw_hh  = (const float*)d_in[12];
  const float* gb_ih  = (const float*)d_in[13];
  const float* gb_hh  = (const float*)d_in[14];
  const float* sw_ih  = (const float*)d_in[15];
  const float* sw_hh  = (const float*)d_in[16];
  const float* sb_ih  = (const float*)d_in[17];
  const float* sb_hh  = (const float*)d_in[18];
  const float* mw_ih  = (const float*)d_in[19];
  const float* mb_ih  = (const float*)d_in[21];
  const float* mb_hh  = (const float*)d_in[22];
  const float* l1_w   = (const float*)d_in[23];
  const float* l1_b   = (const float*)d_in[24];
  const float* l3_w   = (const float*)d_in[25];
  const float* l3_b   = (const float*)d_in[26];
  float* out = (float*)d_out;

  char* ws = (char*)d_ws;
  __bf16* We_bf   = (__bf16*)(ws + WS_WE);
  __bf16* he_bf   = (__bf16*)(ws + WS_HE);
  float*  h_f     = (float*) (ws + WS_H);
  __bf16* h_bf    = (__bf16*)(ws + WS_HBF);
  float*  m_f     = (float*) (ws + WS_M);
  __bf16* m_bf    = (__bf16*)(ws + WS_MBF);
  float*  agg     = (float*) (ws + WS_AGG);
  float*  gi      = (float*) (ws + WS_GI);
  float*  gh      = (float*) (ws + WS_GH);
  float*  deg     = (float*) (ws + WS_DEG);
  float*  small   = (float*) (ws + WS_SMALL);
  __bf16* e2w_bf  = (__bf16*)(ws + WS_E2WBF);
  __bf16* gwih_bf = (__bf16*)(ws + WS_GWIH);
  __bf16* gwhh_bf = (__bf16*)(ws + WS_GWHH);
  __bf16* rootw_bf= (__bf16*)(ws + WS_ROOTW);
  float* qstar = small;          // 128
  float* hs    = small + 128;    // 64
  float* cs    = small + 192;    // 64
  float* racc  = small + 256;    // 64
  float* sumexp= small + 320;    // 1

  const int* src = ei;
  const int* dst = ei + N_EDGES;

  // ---- init (ws is poisoned; zero everything accumulated into) ----
  zero_kernel<<<256, 256, 0, stream>>>(agg, N_NODES * GDIM);
  zero_kernel<<<64, 256, 0, stream>>>(deg, N_NODES);
  zero_kernel<<<1, 256, 0, stream>>>(small, 512);

  // ---- one-time bf16 weight conversions ----
  cvt_bf16_kernel<<<(4096 * 64 + 255) / 256, 256, 0, stream>>>(e2_w, e2w_bf, 4096 * 64);
  cvt_bf16_kernel<<<(192 * 64 + 255) / 256, 256, 0, stream>>>(gw_ih, gwih_bf, 192 * 64);
  cvt_bf16_kernel<<<(192 * 64 + 255) / 256, 256, 0, stream>>>(gw_hh, gwhh_bf, 192 * 64);
  cvt_bf16_t64_kernel<<<(64 * 64 + 255) / 256, 256, 0, stream>>>(root_w, rootw_bf);

  // ---- node / edge embeddings ----
  lin0_kernel<<<(N_NODES * GDIM + 255) / 256, 256, 0, stream>>>(x, lin0_w, lin0_b, h_f, h_bf);
  he_kernel<<<(N_EDGES * GDIM + 255) / 256, 256, 0, stream>>>(eattr, e1_w, e1_b, he_bf);
  deg_kernel<<<(N_EDGES + 255) / 256, 256, 0, stream>>>(dst, deg);

  // ---- edge weight tensor: We = he @ e2_w.T + e2_b  (E x 4096, bf16, WMMA) ----
  gemm_wmma_k64<<<dim3(N_EDGES / 16, 4096 / 64), 32, 0, stream>>>(
      he_bf, e2w_bf, e2_b, nullptr, We_bf, N_EDGES, 4096, 0);

  const int TM = (N_NODES + 15) / 16;  // 938

  // ---- 6 NNConv + GRU steps ----
  for (int step = 0; step < 6; ++step) {
    msg_kernel<<<(N_EDGES * 32 + 255) / 256, 256, 0, stream>>>(src, dst, We_bf, h_f, agg);
    gemm_wmma_k64<<<dim3(TM, 1), 32, 0, stream>>>(
        h_bf, rootw_bf, conv_b, m_f, nullptr, N_NODES, GDIM, 0);
    combine_kernel<<<(N_NODES * GDIM + 255) / 256, 256, 0, stream>>>(deg, m_f, m_bf, agg);
    gemm_wmma_k64<<<dim3(TM, 192 / 64), 32, 0, stream>>>(
        m_bf, gwih_bf, gb_ih, gi, nullptr, N_NODES, 192, 0);
    gemm_wmma_k64<<<dim3(TM, 192 / 64), 32, 0, stream>>>(
        h_bf, gwhh_bf, gb_hh, gh, nullptr, N_NODES, 192, 0);
    gru_kernel<<<(N_NODES * GDIM + 255) / 256, 256, 0, stream>>>(gi, gh, h_f, h_bf);
  }

  // ---- Set2Set pooling (6 rounds) ----
  for (int it = 0; it < 6; ++it) {
    s2s_lstm_kernel<<<1, 256, 0, stream>>>(sw_ih, sw_hh, sb_ih, sb_hh, qstar, hs, cs,
                                           racc, sumexp);
    attn_kernel<<<240, 256, 0, stream>>>(h_f, hs, racc, sumexp);
    s2s_finalize_kernel<<<1, 64, 0, stream>>>(hs, racc, sumexp, qstar);
  }

  // ---- head ----
  head_kernel<<<1, 256, 0, stream>>>(qstar, mw_ih, mb_ih, mb_hh, l1_w, l1_b, l3_w, l3_b, out);
}
